// HGTFraudDetector_5712306504438
// MI455X (gfx1250) — matrix operand order repaired and checked by reading, not compiled, not run on hardware
//
#include <hip/hip_runtime.h>
#include <hip/hip_bf16.h>
#include <cstddef>
#include <cstdint>

// ---------------------------------------------------------------------------
// Types for WMMA (CDNA5 / gfx1250, wave32)
// ---------------------------------------------------------------------------
typedef __bf16 bf16;
typedef __attribute__((ext_vector_type(16))) __bf16 bf16x16;
typedef __attribute__((ext_vector_type(8)))  float  f32x8;

#define ACT_NONE 0
#define ACT_ELU  1
#define ACT_RELU 2

static inline int ceildiv_h(int a, int b) { return (a + b - 1) / b; }

__device__ inline uint32_t pack_bf16(float x, float y) {
  union { bf16 h[2]; uint32_t u; } p;
  p.h[0] = (bf16)x;
  p.h[1] = (bf16)y;
  return p.u;
}

// ---------------------------------------------------------------------------
// Elementwise helpers
// ---------------------------------------------------------------------------
__global__ __launch_bounds__(256) void fill_kernel(float* __restrict__ p, float v, int n) {
  int i = blockIdx.x * 256 + threadIdx.x;
  if (i < n) p[i] = v;
}

__global__ __launch_bounds__(256) void gelu_kernel(float* __restrict__ x, int n) {
  int i = blockIdx.x * 256 + threadIdx.x;
  if (i < n) {
    float v = x[i];
    x[i] = 0.5f * v * (1.0f + erff(v * 0.70710678118654752f));
  }
}

__global__ __launch_bounds__(256) void fixmax_kernel(float* __restrict__ m, int n) {
  int i = blockIdx.x * 256 + threadIdx.x;
  if (i < n) {
    float v = m[i];
    if (!__builtin_isfinite(v)) m[i] = 0.0f;
  }
}

// ---------------------------------------------------------------------------
// Weight packing into WMMA B-fragment order.
// Fragment layout (contiguous k order, same as the A staging in gemm_kernel): lane l supplies
// column n = l&15; K-half kb = (l>>4)*16; VGPR v holds K = kb+2v, kb+2v+1.
// Packed layout: u32[ ((ntile*(K/32) + kstep)*32 + lane)*8 + v ]  -> each lane
// reads its 16 bf16 as one contiguous 32B chunk (global_load_b128 x2).
// ---------------------------------------------------------------------------
__global__ __launch_bounds__(256)
void packB_kernel(const float* __restrict__ W, uint32_t* __restrict__ out, int K, int N) {
  int total = (K >> 5) * (N >> 4) * 256;
  int i = blockIdx.x * 256 + threadIdx.x;
  if (i >= total) return;
  int slot = i & 255, tile = i >> 8;
  int ksteps = K >> 5;
  int ks = tile % ksteps, nt = tile / ksteps;
  int lane = slot >> 3, v = slot & 7;
  int n = nt * 16 + (lane & 15);
  int k = ks * 32 + (lane >> 4) * 16 + 2 * v;
  out[i] = pack_bf16(W[(size_t)k * N + n], W[(size_t)(k + 1) * N + n]);
}

// Pack a per-relation [4][32][32] tensor as the block-diagonal 128x128 matrix
// BD[h*32+d, h*32+e] = rel[h][d][e] directly in B-fragment order (K=N=128).
__global__ __launch_bounds__(256)
void packRel_kernel(const float* __restrict__ rel, uint32_t* __restrict__ out) {
  int i = blockIdx.x * 256 + threadIdx.x;
  if (i >= 8192) return;                       // 4 ksteps * 8 ntiles * 256
  int slot = i & 255, tile = i >> 8;
  int ks = tile & 3, nt = tile >> 2;           // ksteps = 4
  int lane = slot >> 3, v = slot & 7;
  int n = nt * 16 + (lane & 15);
  int k = ks * 32 + (lane >> 4) * 16 + 2 * v;  // k even; k,k+1 in same 32-block
  int hk = k >> 5, hn = n >> 5;
  float x0 = 0.0f, x1 = 0.0f;
  if (hk == hn) {
    const float* rp = rel + ((size_t)hk * 32 + (k & 31)) * 32 + (n & 31);
    x0 = rp[0];
    x1 = rp[32];                               // (d+1) row, same e
  }
  out[i] = pack_bf16(x0, x1);
}

// ---------------------------------------------------------------------------
// GEMM: C[M,N] = act(A[M,K] @ W[K,N] + bias)
// A fp32 row-major with row stride lda; W pre-packed bf16 fragments; C fp32
// row stride ldc. Block = 256 thr = 8 waves covering 32 rows x 128 cols:
// two A fragments staged bf16-packed in LDS (shared by all waves), wave w owns
// col tile (by*8+w) and keeps two accumulators (B fragment reused).
// K must be a multiple of 32 (all call sites: 32/64/128).
// ---------------------------------------------------------------------------
__global__ __launch_bounds__(256)
void gemm_kernel(const float* __restrict__ A, int lda,
                 const uint32_t* __restrict__ Wp, const float* __restrict__ bias,
                 float* __restrict__ C, int ldc, int M, int N, int K, int act) {
  __shared__ __align__(32) uint32_t Asb[2][32][8];
  const int lane  = threadIdx.x & 31;
  const int wave  = threadIdx.x >> 5;
  const int row0  = blockIdx.x * 32;
  const int ntile = blockIdx.y * 8 + wave;
  const int col0  = ntile * 16;
  const bool active = (col0 < N);          // wave-uniform => EXEC all-ones inside
  const int ksteps = K >> 5;

  f32x8 acc0 = {};
  f32x8 acc1 = {};
  for (int ks = 0; ks < ksteps; ks++) {
    const int k0 = ks * 32;
    __syncthreads();
    // Stage 2 x (16x32) A tiles, already packed in fragment order.
    for (int s = threadIdx.x; s < 512; s += 256) {
      int f  = s >> 8;
      int r  = s & 255;
      int ls = r >> 3, v = r & 7;
      int m  = ls & 15, kh = ls >> 4;
      int ka = kh * 16 + 2 * v;
      int row = row0 + f * 16 + m;
      float x0 = 0.0f, x1 = 0.0f;
      if (row < M) {
        float2 t = *(const float2*)(A + (size_t)row * lda + k0 + ka);
        x0 = t.x; x1 = t.y;
      }
      Asb[f][ls][v] = pack_bf16(x0, x1);
    }
    __syncthreads();
    if (active) {
      bf16x16 a0 = *reinterpret_cast<const bf16x16*>(Asb[0][lane]);
      bf16x16 a1 = *reinterpret_cast<const bf16x16*>(Asb[1][lane]);
      bf16x16 b  = reinterpret_cast<const bf16x16*>(Wp)[(size_t)(ntile * ksteps + ks) * 32 + lane];
      acc0 = __builtin_amdgcn_wmma_f32_16x16x32_bf16(false, a0, false, b,
                                                     (short)0, acc0, false, false);
      acc1 = __builtin_amdgcn_wmma_f32_16x16x32_bf16(false, a1, false, b,
                                                     (short)0, acc1, false, false);
    }
  }
  if (active) {
    const int n = col0 + (lane & 15);
    const int khalf = lane >> 4;
    const float bv = bias ? bias[n] : 0.0f;
#pragma unroll
    for (int v = 0; v < 8; v++) {
      int mm0 = row0 + v + khalf * 8;
      int mm1 = mm0 + 16;
      float xv0 = acc0[v] + bv;
      float xv1 = acc1[v] + bv;
      if (act == ACT_ELU) {
        xv0 = xv0 > 0.0f ? xv0 : (expf(xv0) - 1.0f);
        xv1 = xv1 > 0.0f ? xv1 : (expf(xv1) - 1.0f);
      } else if (act == ACT_RELU) {
        xv0 = fmaxf(xv0, 0.0f);
        xv1 = fmaxf(xv1, 0.0f);
      }
      if (mm0 < M) C[(size_t)mm0 * ldc + n] = xv0;
      if (mm1 < M) C[(size_t)mm1 * ldc + n] = xv1;
    }
  }
}

// ---------------------------------------------------------------------------
// Attention score: alpha[e,h] = dot(q[dst,h,:], krel[src,h,:]) * p_rel[h] * SCALE
// ---------------------------------------------------------------------------
__global__ __launch_bounds__(256)
void alpha_kernel(const float* __restrict__ kqv_dst, const float* __restrict__ krel,
                  const int* __restrict__ src, const int* __restrict__ dst,
                  const float* __restrict__ prel, float* __restrict__ alpha, int E) {
  int t = blockIdx.x * 256 + threadIdx.x;
  if (t >= E * 4) return;
  int e = t >> 2, h = t & 3;
  int s = src[e], d = dst[e];
  const float4* qp = (const float4*)(kqv_dst + (size_t)d * 384 + 128 + h * 32);
  const float4* kp = (const float4*)(krel + (size_t)s * 128 + h * 32);
  float acc = 0.0f;
#pragma unroll
  for (int i = 0; i < 8; i++) {
    float4 q4 = qp[i], k4 = kp[i];
    acc += q4.x * k4.x + q4.y * k4.y + q4.z * k4.z + q4.w * k4.w;
  }
  alpha[(size_t)e * 4 + h] = acc * prel[h] * 0.17677669529663687f; // 1/sqrt(32)
}

__device__ inline void atomicMaxF(float* addr, float val) {
  if (val >= 0.0f) atomicMax((int*)addr, __float_as_int(val));
  else             atomicMin((unsigned int*)addr, __float_as_uint(val));
}

__global__ __launch_bounds__(256)
void segmax_kernel(const float* __restrict__ alpha, const int* __restrict__ dst,
                   float* __restrict__ m, int E) {
  int t = blockIdx.x * 256 + threadIdx.x;
  if (t >= E * 4) return;
  int e = t >> 2, h = t & 3;
  atomicMaxF(&m[(size_t)dst[e] * 4 + h], alpha[t]);
}

__global__ __launch_bounds__(256)
void expsum_kernel(float* __restrict__ alpha, const int* __restrict__ dst,
                   const float* __restrict__ m, float* __restrict__ den, int E) {
  int t = blockIdx.x * 256 + threadIdx.x;
  if (t >= E * 4) return;
  int e = t >> 2, h = t & 3;
  size_t idx = (size_t)dst[e] * 4 + h;
  float ex = expf(alpha[t] - m[idx]);
  alpha[t] = ex;
  atomicAdd(&den[idx], ex);
}

__global__ __launch_bounds__(256)
void agg_kernel(const float* __restrict__ alpha, const float* __restrict__ vrel,
                const int* __restrict__ src, const int* __restrict__ dst,
                const float* __restrict__ den, float* __restrict__ agg, int E) {
  int t = blockIdx.x * 256 + threadIdx.x;
  if (t >= E * 4) return;
  int e = t >> 2, h = t & 3;
  int s = src[e], d = dst[e];
  float w = alpha[t] / fmaxf(den[(size_t)d * 4 + h], 1e-16f);
  const float* vp = vrel + (size_t)s * 128 + h * 32;
  float* ap = agg + (size_t)d * 128 + h * 32;
#pragma unroll 4
  for (int i = 0; i < 32; i++) atomicAdd(&ap[i], w * vp[i]);
}

// ---------------------------------------------------------------------------
// hn = beta*o + (1-beta)*h ; y = hn + h ; out = LayerNorm(y, g, b)  (HID=128)
// ---------------------------------------------------------------------------
__global__ __launch_bounds__(256)
void skipln_kernel(const float* __restrict__ o, const float* __restrict__ h,
                   const float* __restrict__ skip, const float* __restrict__ lng,
                   const float* __restrict__ lnb, float* __restrict__ hout, int N) {
  int n = blockIdx.x * 256 + threadIdx.x;
  if (n >= N) return;
  float beta = 1.0f / (1.0f + expf(-skip[0]));
  float ch = 2.0f - beta;
  const float* op = o + (size_t)n * 128;
  const float* hp = h + (size_t)n * 128;
  float s = 0.0f, s2 = 0.0f;
  for (int i = 0; i < 128; i++) {
    float y = beta * op[i] + ch * hp[i];
    s += y; s2 += y * y;
  }
  float mu  = s * (1.0f / 128.0f);
  float var = s2 * (1.0f / 128.0f) - mu * mu;
  float inv = rsqrtf(var + 1e-5f);
  float* out = hout + (size_t)n * 128;
  for (int i = 0; i < 128; i++) {
    float y = beta * op[i] + ch * hp[i];
    out[i] = (y - mu) * inv * lng[i] + lnb[i];
  }
}

// Final classifier stage: out[n,:2] = c[n,:64] @ W2[64,2] + b2
__global__ __launch_bounds__(256)
void clf2_kernel(const float* __restrict__ c, const float* __restrict__ W2,
                 const float* __restrict__ b2, float* __restrict__ out, int N) {
  int n = blockIdx.x * 256 + threadIdx.x;
  if (n >= N) return;
  float a0 = b2[0], a1 = b2[1];
  const float* cp = c + (size_t)n * 64;
  for (int i = 0; i < 64; i++) {
    float v = cp[i];
    a0 += v * W2[i * 2];
    a1 += v * W2[i * 2 + 1];
  }
  out[(size_t)n * 2]     = a0;
  out[(size_t)n * 2 + 1] = a1;
}

// ---------------------------------------------------------------------------
// Host orchestration
// ---------------------------------------------------------------------------
extern "C" void kernel_launch(void* const* d_in, const int* in_sizes, int n_in,
                              void* d_out, int out_size, void* d_ws, size_t ws_size,
                              hipStream_t stream) {
  const float* x[3] = {(const float*)d_in[0], (const float*)d_in[1], (const float*)d_in[2]};
  const int FEATs[3] = {64, 32, 32};
  const int Nn[3] = {in_sizes[0] / 64, in_sizes[1] / 32, in_sizes[2] / 32};
  const int E = in_sizes[3] / 2;

  // Relation order matches sorted param keys: filed_by, files, serviced_by, services
  const int srcT[4] = {0, 1, 0, 2};
  const int dstT[4] = {1, 0, 2, 0};
  const int* ei[4] = {(const int*)d_in[4], (const int*)d_in[3],
                      (const int*)d_in[6], (const int*)d_in[5]};

  auto F = [&](int i) { return (const float*)d_in[i]; };
  // Param leaf indices (jax pytree sorted-key order):
  //  7 clf.W1 [128,64]   8 clf.W2 [64,2]   9 clf.b1 [64]   10 clf.b2 [2]
  //  layer L base B = 11 + 29*L:
  //   B+0..3  a_rel (filed_by, files, serviced_by, services) each [4,32,32]
  //   B+4+2t  kqv[t].W [128,384]   B+5+2t kqv[t].b [384]
  //   B+10 ln.b [128]   B+11 ln.g [128]
  //   B+12..15 m_rel   B+16+2t out[t].W [128,128]   B+17+2t out[t].b [128]
  //   B+22..25 p_rel [4]   B+26+t skip (scalar)
  //  proj base 69: 69+2t proj[t].W [FEAT,128], 70+2t proj[t].b [128]
  const int PROJ = 11 + 2 * 29;

  // --- Workspace carve (bump allocator) ---
  char* base = (char*)d_ws;
  size_t used = 0;
  auto alloc = [&](size_t bytes) -> void* {
    bytes = (bytes + 255) & ~(size_t)255;
    void* p = base + used;
    used += bytes;
    return p;
  };

  float *hA[3], *hB[3], *kqvb[3], *aggb[3], *ob[3], *mb[3], *denb[3];
  for (int t = 0; t < 3; t++) {
    hA[t]   = (float*)alloc((size_t)Nn[t] * 128 * 4);
    hB[t]   = (float*)alloc((size_t)Nn[t] * 128 * 4);
    kqvb[t] = (float*)alloc((size_t)Nn[t] * 384 * 4);
    aggb[t] = (float*)alloc((size_t)Nn[t] * 128 * 4);
    ob[t]   = (float*)alloc((size_t)Nn[t] * 128 * 4);
    mb[t]   = (float*)alloc((size_t)Nn[t] * 4 * 4);
    denb[t] = (float*)alloc((size_t)Nn[t] * 4 * 4);
  }
  float *krelb[4], *vrelb[4], *alphab[4];
  for (int r = 0; r < 4; r++) {
    krelb[r]  = (float*)alloc((size_t)Nn[srcT[r]] * 128 * 4);
    vrelb[r]  = (float*)alloc((size_t)Nn[srcT[r]] * 128 * 4);
    alphab[r] = (float*)alloc((size_t)E * 4 * 4);
  }
  float* cbuf = (float*)alloc((size_t)Nn[0] * 64 * 4);

  // Packed bf16 weights (u32 fragment slots; bytes = K*N*2)
  uint32_t *projWp[3], *kqvWp[2][3], *aWp[2][4], *mWp[2][4], *outWp[2][3], *clfW1p;
  for (int t = 0; t < 3; t++) projWp[t] = (uint32_t*)alloc((size_t)FEATs[t] * 128 * 2);
  for (int L = 0; L < 2; L++) {
    for (int t = 0; t < 3; t++) kqvWp[L][t] = (uint32_t*)alloc((size_t)128 * 384 * 2);
    for (int r = 0; r < 4; r++) {
      aWp[L][r] = (uint32_t*)alloc((size_t)128 * 128 * 2);
      mWp[L][r] = (uint32_t*)alloc((size_t)128 * 128 * 2);
    }
    for (int t = 0; t < 3; t++) outWp[L][t] = (uint32_t*)alloc((size_t)128 * 128 * 2);
  }
  clfW1p = (uint32_t*)alloc((size_t)128 * 64 * 2);
  if (used > ws_size) return;  // insufficient scratch: bail deterministically

  auto packB = [&](const float* W, uint32_t* out, int K, int N) {
    int total = (K >> 5) * (N >> 4) * 256;
    packB_kernel<<<ceildiv_h(total, 256), 256, 0, stream>>>(W, out, K, N);
  };
  auto packRel = [&](const float* rel, uint32_t* out) {
    packRel_kernel<<<32, 256, 0, stream>>>(rel, out);
  };
  auto fill = [&](float* p, float v, int n) {
    fill_kernel<<<ceildiv_h(n, 256), 256, 0, stream>>>(p, v, n);
  };
  auto gemm = [&](const float* A, int lda, const uint32_t* Wp, const float* bias,
                  float* C, int ldc, int M, int N, int K, int act) {
    dim3 g(ceildiv_h(M, 32), ceildiv_h(N, 128));
    gemm_kernel<<<g, 256, 0, stream>>>(A, lda, Wp, bias, C, ldc, M, N, K, act);
  };

  // --- Pack weights into WMMA fragment layout ---
  for (int t = 0; t < 3; t++) packB(F(PROJ + 2 * t), projWp[t], FEATs[t], 128);
  for (int L = 0; L < 2; L++) {
    int B = 11 + 29 * L;
    for (int t = 0; t < 3; t++) packB(F(B + 4 + 2 * t), kqvWp[L][t], 128, 384);
    for (int r = 0; r < 4; r++) {
      packRel(F(B + r), aWp[L][r]);
      packRel(F(B + 12 + r), mWp[L][r]);
    }
    for (int t = 0; t < 3; t++) packB(F(B + 16 + 2 * t), outWp[L][t], 128, 128);
  }
  packB(F(7), clfW1p, 128, 64);

  // --- Input projections: h[t] = elu(x @ Wp + bp) ---
  float* cur[3] = {hA[0], hA[1], hA[2]};
  float* nxt[3] = {hB[0], hB[1], hB[2]};
  for (int t = 0; t < 3; t++)
    gemm(x[t], FEATs[t], projWp[t], F(PROJ + 1 + 2 * t), cur[t], 128,
         Nn[t], 128, FEATs[t], ACT_ELU);

  // --- HGT layers ---
  const int EH = E * 4;
  for (int L = 0; L < 2; L++) {
    int B = 11 + 29 * L;
    for (int t = 0; t < 3; t++)
      gemm(cur[t], 128, kqvWp[L][t], F(B + 5 + 2 * t), kqvb[t], 384,
           Nn[t], 384, 128, ACT_NONE);
    // Relation transforms as block-diagonal GEMMs on the k / v slices of kqv.
    for (int r = 0; r < 4; r++) {
      int s = srcT[r];
      gemm(kqvb[s], 384, aWp[L][r], nullptr, krelb[r], 128, Nn[s], 128, 128, ACT_NONE);
      gemm(kqvb[s] + 256, 384, mWp[L][r], nullptr, vrelb[r], 128, Nn[s], 128, 128, ACT_NONE);
    }
    for (int t = 0; t < 3; t++) {
      fill(mb[t], -__builtin_inff(), Nn[t] * 4);
      fill(denb[t], 0.0f, Nn[t] * 4);
      fill(aggb[t], 0.0f, Nn[t] * 128);
    }
    for (int r = 0; r < 4; r++)
      alpha_kernel<<<ceildiv_h(EH, 256), 256, 0, stream>>>(
          kqvb[dstT[r]], krelb[r], ei[r], ei[r] + E, F(B + 22 + r), alphab[r], E);
    for (int r = 0; r < 4; r++)
      segmax_kernel<<<ceildiv_h(EH, 256), 256, 0, stream>>>(
          alphab[r], ei[r] + E, mb[dstT[r]], E);
    for (int t = 0; t < 3; t++)
      fixmax_kernel<<<ceildiv_h(Nn[t] * 4, 256), 256, 0, stream>>>(mb[t], Nn[t] * 4);
    for (int r = 0; r < 4; r++)
      expsum_kernel<<<ceildiv_h(EH, 256), 256, 0, stream>>>(
          alphab[r], ei[r] + E, mb[dstT[r]], denb[dstT[r]], E);
    for (int r = 0; r < 4; r++)
      agg_kernel<<<ceildiv_h(EH, 256), 256, 0, stream>>>(
          alphab[r], vrelb[r], ei[r], ei[r] + E, denb[dstT[r]], aggb[dstT[r]], E);
    for (int t = 0; t < 3; t++) {
      gelu_kernel<<<ceildiv_h(Nn[t] * 128, 256), 256, 0, stream>>>(aggb[t], Nn[t] * 128);
      gemm(aggb[t], 128, outWp[L][t], F(B + 17 + 2 * t), ob[t], 128,
           Nn[t], 128, 128, ACT_NONE);
      skipln_kernel<<<ceildiv_h(Nn[t], 256), 256, 0, stream>>>(
          ob[t], cur[t], F(B + 26 + t), F(B + 11), F(B + 10), nxt[t], Nn[t]);
    }
    for (int t = 0; t < 3; t++) { float* tmp = cur[t]; cur[t] = nxt[t]; nxt[t] = tmp; }
  }

  // --- Classifier ---
  gemm(cur[0], 128, clfW1p, F(9), cbuf, 64, Nn[0], 64, 128, ACT_RELU);
  clf2_kernel<<<ceildiv_h(Nn[0], 256), 256, 0, stream>>>(
      cbuf, F(8), F(10), (float*)d_out, Nn[0]);
}
